// LSTM_81810537055149
// MI455X (gfx1250) — matrix operand order repaired and checked by reading, not compiled, not run on hardware
//
#include <hip/hip_runtime.h>
#include <hip/hip_bf16.h>

// ---------------------------------------------------------------------------
// Types for CDNA5 WMMA (wave32, V_WMMA_F32_16X16X32_BF16)
// ---------------------------------------------------------------------------
typedef __attribute__((ext_vector_type(16))) __bf16 v16bf;
typedef __attribute__((ext_vector_type(8)))  __bf16 v8bf;
typedef __attribute__((ext_vector_type(8)))  float  v8f;

#define EPS 1e-5f

__device__ __forceinline__ unsigned short f2bf_bits(float f) {
    unsigned u = __builtin_bit_cast(unsigned, f);
    unsigned r = u + 0x7FFFu + ((u >> 16) & 1u);   // round-to-nearest-even
    return (unsigned short)(r >> 16);
}
__device__ __forceinline__ float bf2f(unsigned short s) {
    return __builtin_bit_cast(float, ((unsigned)s) << 16);
}
__device__ __forceinline__ float sigm(float x) { return 1.0f / (1.0f + expf(-x)); }

__device__ __forceinline__ v16bf cat16(v8bf lo, v8bf hi) {
    return __builtin_shufflevector(lo, hi, 0, 1, 2, 3, 4, 5, 6, 7,
                                           8, 9, 10, 11, 12, 13, 14, 15);
}

// ---------------------------------------------------------------------------
// Elementwise convert f32 -> bf16 (row-major preserved)
// ---------------------------------------------------------------------------
__global__ void cvt_f32_bf16(const float* __restrict__ in, unsigned short* __restrict__ out, size_t n) {
    size_t i = (size_t)blockIdx.x * blockDim.x + threadIdx.x;
    if (i < n) out[i] = f2bf_bits(in[i]);
}

// Transpose + convert: W (K x N, f32) -> Wt (N x K, bf16). One-time cost.
__global__ void transpose_f32_bf16(const float* __restrict__ W, unsigned short* __restrict__ Wt,
                                   int K, int N) {
    size_t i = (size_t)blockIdx.x * blockDim.x + threadIdx.x;
    if (i >= (size_t)K * N) return;
    int k = (int)(i / N), n = (int)(i % N);
    Wt[(size_t)n * K + k] = f2bf_bits(W[i]);
}

__global__ void zero_state(float* __restrict__ h, float* __restrict__ c,
                           unsigned short* __restrict__ hb, int n) {
    int i = blockIdx.x * blockDim.x + threadIdx.x;
    if (i < n) { h[i] = 0.0f; c[i] = 0.0f; hb[i] = 0; }
}

// ---------------------------------------------------------------------------
// WMMA GEMM:  C[M x N] = A[M x K] * Bt[N x K]^T   (A,Bt bf16, accum f32)
// Each wave owns a 16x64 output strip (A-fragment reused over 4 B tiles).
// SPLIT-K: with split>1 the grid carries split x strips; k-part p covers
// K-range [p*K/split, (p+1)*K/split) and writes partial sums to
// C + p*M*N (consumer adds the partials). Doubles wave-parallelism for the
// latency-critical 64x4096 recurrent GEMM.
// K loop is software-pipelined with double-buffered fragment registers so
// global_load_b128s for chunk k+32 overlap v_wmma of chunk k.
// Requires (K/split) % 64 == 0.  A/B: contiguous K per lane half (same order for both); C/D per CDNA5 ISA 7.12.2.
// ---------------------------------------------------------------------------
template <bool OUT_BF16>
__global__ void gemm_wmma_bf16(const unsigned short* __restrict__ A,
                               const unsigned short* __restrict__ Bt,
                               void* __restrict__ Cv, int M, int N, int K, int split) {
    const int lane  = threadIdx.x & 31;
    const int wave  = threadIdx.x >> 5;
    int strip = blockIdx.x * (blockDim.x >> 5) + wave;
    const int stripsN = N >> 6;                       // strips of 64 cols
    const int nStrips = (M >> 4) * stripsN;
    if (strip >= nStrips * split) return;
    const int kpart = strip / nStrips;
    strip -= kpart * nStrips;
    const int Klen  = K / split;
    const int kbase = kpart * Klen;
    const int tm = strip / stripsN;
    const int n0 = (strip - tm * stripsN) << 6;
    const int m0 = tm << 4;
    const int r    = lane & 15;
    const int half = lane >> 4;

    // A: lane<16 holds row m0+r, K(0..15) contiguous; lane>=16: K(16..31)
    const unsigned short* pa = A + (size_t)(m0 + r) * K + kbase + half * 16;
    // B: lane<16 holds column n0+r, K(0..15) contiguous; lane>=16: K(16..31)
    const unsigned short* pb0 = Bt + (size_t)(n0 + r) * K + kbase + half * 16;
    const unsigned short* pb1 = pb0 + (size_t)16 * K;
    const unsigned short* pb2 = pb0 + (size_t)32 * K;
    const unsigned short* pb3 = pb0 + (size_t)48 * K;

    v8f acc[4] = {};

    // double-buffered fragments
    v8bf aLo0, aHi0, aLo1, aHi1;
    v8bf bLo0[4], bHi0[4], bLo1[4], bHi1[4];

    // prologue: chunk 0 -> buffer 0
    aLo0 = *(const v8bf*)(pa);
    aHi0 = *(const v8bf*)(pa + 8);
    bLo0[0] = *(const v8bf*)(pb0); bHi0[0] = *(const v8bf*)(pb0 + 8);
    bLo0[1] = *(const v8bf*)(pb1); bHi0[1] = *(const v8bf*)(pb1 + 8);
    bLo0[2] = *(const v8bf*)(pb2); bHi0[2] = *(const v8bf*)(pb2 + 8);
    bLo0[3] = *(const v8bf*)(pb3); bHi0[3] = *(const v8bf*)(pb3 + 8);

    for (int k0 = 0; k0 < Klen; k0 += 64) {
        // prefetch chunk k0+32 -> buffer 1 (always valid: Klen % 64 == 0)
        {
            const int kq = k0 + 32;
            aLo1 = *(const v8bf*)(pa + kq);
            aHi1 = *(const v8bf*)(pa + kq + 8);
            bLo1[0] = *(const v8bf*)(pb0 + kq); bHi1[0] = *(const v8bf*)(pb0 + kq + 8);
            bLo1[1] = *(const v8bf*)(pb1 + kq); bHi1[1] = *(const v8bf*)(pb1 + kq + 8);
            bLo1[2] = *(const v8bf*)(pb2 + kq); bHi1[2] = *(const v8bf*)(pb2 + kq + 8);
            bLo1[3] = *(const v8bf*)(pb3 + kq); bHi1[3] = *(const v8bf*)(pb3 + kq + 8);
        }
        // compute chunk k0 from buffer 0
        {
            v16bf a = cat16(aLo0, aHi0);
#pragma unroll
            for (int tn = 0; tn < 4; ++tn)
                acc[tn] = __builtin_amdgcn_wmma_f32_16x16x32_bf16(
                    false, a, false, cat16(bLo0[tn], bHi0[tn]), (short)0, acc[tn],
                    false, false);
        }
        // prefetch chunk k0+64 -> buffer 0 (while buffer-1 computes)
        if (k0 + 64 < Klen) {
            const int kq = k0 + 64;
            aLo0 = *(const v8bf*)(pa + kq);
            aHi0 = *(const v8bf*)(pa + kq + 8);
            bLo0[0] = *(const v8bf*)(pb0 + kq); bHi0[0] = *(const v8bf*)(pb0 + kq + 8);
            bLo0[1] = *(const v8bf*)(pb1 + kq); bHi0[1] = *(const v8bf*)(pb1 + kq + 8);
            bLo0[2] = *(const v8bf*)(pb2 + kq); bHi0[2] = *(const v8bf*)(pb2 + kq + 8);
            bLo0[3] = *(const v8bf*)(pb3 + kq); bHi0[3] = *(const v8bf*)(pb3 + kq + 8);
        }
        // compute chunk k0+32 from buffer 1
        {
            v16bf a = cat16(aLo1, aHi1);
#pragma unroll
            for (int tn = 0; tn < 4; ++tn)
                acc[tn] = __builtin_amdgcn_wmma_f32_16x16x32_bf16(
                    false, a, false, cat16(bLo1[tn], bHi1[tn]), (short)0, acc[tn],
                    false, false);
        }
    }

    // C/D f32 16x16: VGPR i, lanes 0-15 -> M=i, N=lane; lanes 16-31 -> M=8+i
    const size_t cOff = (size_t)kpart * M * N;
#pragma unroll
    for (int tn = 0; tn < 4; ++tn) {
#pragma unroll
        for (int i = 0; i < 8; ++i) {
            const size_t mm = (size_t)(m0 + i + half * 8);
            const size_t nn = (size_t)(n0 + tn * 16 + r);
            if (OUT_BF16) ((unsigned short*)Cv)[cOff + mm * N + nn] = f2bf_bits(acc[tn][i]);
            else          ((float*)Cv)[cOff + mm * N + nn]          = acc[tn][i];
        }
    }
}

// ---------------------------------------------------------------------------
// In-place BN over the batch dim for precomputed input projections.
// S: (T, B, C) bf16. One thread per (t, col): stats over B=64, normalize.
// ---------------------------------------------------------------------------
__global__ void bn_cols_bf16(unsigned short* __restrict__ S,
                             const float* __restrict__ g, const float* __restrict__ bt,
                             int B, int C, size_t total /* T*C */) {
    size_t idx = (size_t)blockIdx.x * blockDim.x + threadIdx.x;
    if (idx >= total) return;
    size_t t = idx / C;
    int n = (int)(idx % C);
    unsigned short* p = S + t * (size_t)B * C + n;
    float sum = 0.f, sq = 0.f;
    for (int b = 0; b < B; ++b) {
        float v = bf2f(p[(size_t)b * C]);
        sum += v; sq += v * v;
    }
    float invB = 1.0f / (float)B;
    float mean = sum * invB;
    float var  = fmaxf(sq * invB - mean * mean, 0.0f);
    float inv  = rsqrtf(var + EPS);
    float a  = g[n] * inv;
    float bb = bt[n] - a * mean;
    for (int b = 0; b < B; ++b) {
        size_t o = (size_t)b * C;
        p[o] = f2bf_bits(a * bf2f(p[o]) + bb);
    }
}

// ---------------------------------------------------------------------------
// Per-timestep pointwise, parallel version.
// Block = 256 threads = 4 batch-groups x 64 columns; grid = H/64 blocks.
// s_hh comes as TWO split-K partials that are summed on read.
// Phase 1: BN stats of the 4 gate columns (16-row partials + LDS reduction).
// Phase 2: raw c1 into LDS tile + BN(c) stats (LDS reduction).
// Phase 3: h update, length masking, outputs.
// ---------------------------------------------------------------------------
__global__ void lstm_pointwise(const float* __restrict__ sA,            // B x 4H f32 (k-part 0)
                               const float* __restrict__ sB,            // B x 4H f32 (k-part 1)
                               const unsigned short* __restrict__ bn_ih,// B x 4H bf16 (slice at t)
                               const float* __restrict__ bias,          // 4H
                               const float* __restrict__ g_hh, const float* __restrict__ bt_hh,
                               const float* __restrict__ g_c,  const float* __restrict__ bt_c,
                               const int* __restrict__ length, int t, int B, int H,
                               float* __restrict__ h, float* __restrict__ c,
                               unsigned short* __restrict__ hb,
                               float* __restrict__ y32, unsigned short* __restrict__ y16) {
    const int tx   = threadIdx.x;
    const int col  = tx & 63;          // column within block
    const int bgrp = tx >> 6;          // batch group: rows [16*bgrp, 16*bgrp+16)
    const int b0   = bgrp * 16;
    const int j    = blockIdx.x * 64 + col;
    const int C4   = 4 * H;
    const int cf = j, ci = H + j, co = 2 * H + j, cg = 3 * H + j;
    const float invB = 1.0f / (float)B;

    __shared__ float red[8 * 256];     // reduction scratch
    __shared__ float coef[10 * 64];    // af,bf,ai,bi,ao,bo,ag,bg,ac,bc per column
    __shared__ float craw[64 * 64];    // raw c1 tile [b][col]

    // ---- phase 1: per-group partial stats of the 4 gate columns
    float p0=0,q0=0,p1=0,q1=0,p2=0,q2=0,p3=0,q3=0;
    for (int bb = 0; bb < 16; ++bb) {
        const size_t ro = (size_t)(b0 + bb) * C4;
        float vf = sA[ro + cf] + sB[ro + cf];
        float vi = sA[ro + ci] + sB[ro + ci];
        float vo = sA[ro + co] + sB[ro + co];
        float vg = sA[ro + cg] + sB[ro + cg];
        p0+=vf; q0+=vf*vf; p1+=vi; q1+=vi*vi; p2+=vo; q2+=vo*vo; p3+=vg; q3+=vg*vg;
    }
    red[0*256+tx]=p0; red[1*256+tx]=q0; red[2*256+tx]=p1; red[3*256+tx]=q1;
    red[4*256+tx]=p2; red[5*256+tx]=q2; red[6*256+tx]=p3; red[7*256+tx]=q3;
    __syncthreads();
    if (tx < 64) {
        const int jj = blockIdx.x * 64 + tx;
        const int cols[4] = { jj, H + jj, 2 * H + jj, 3 * H + jj };
#pragma unroll
        for (int gidx = 0; gidx < 4; ++gidx) {
            float s = red[(2*gidx+0)*256+tx]   + red[(2*gidx+0)*256+64+tx]
                    + red[(2*gidx+0)*256+128+tx] + red[(2*gidx+0)*256+192+tx];
            float q = red[(2*gidx+1)*256+tx]   + red[(2*gidx+1)*256+64+tx]
                    + red[(2*gidx+1)*256+128+tx] + red[(2*gidx+1)*256+192+tx];
            float m  = s * invB;
            float iv = rsqrtf(fmaxf(q * invB - m * m, 0.f) + EPS);
            int   cc = cols[gidx];
            float a  = g_hh[cc] * iv;
            coef[(2*gidx+0)*64+tx] = a;
            coef[(2*gidx+1)*64+tx] = bt_hh[cc] - a * m + bias[cc];
        }
    }
    __syncthreads();
    const float af = coef[0*64+col], bf_ = coef[1*64+col];
    const float ai = coef[2*64+col], bi_ = coef[3*64+col];
    const float ao = coef[4*64+col], bo_ = coef[5*64+col];
    const float ag = coef[6*64+col], bg_ = coef[7*64+col];

    // ---- phase 2: raw c1 into LDS + partial BN(c) stats
    float cs = 0.f, cq = 0.f;
    for (int bb = 0; bb < 16; ++bb) {
        const int b = b0 + bb;
        const size_t ro = (size_t)b * C4;
        float sf = af * (sA[ro + cf] + sB[ro + cf]) + bf_ + bf2f(bn_ih[ro + cf]);
        float si = ai * (sA[ro + ci] + sB[ro + ci]) + bi_ + bf2f(bn_ih[ro + ci]);
        float sg = ag * (sA[ro + cg] + sB[ro + cg]) + bg_ + bf2f(bn_ih[ro + cg]);
        float c1 = sigm(sf) * c[(size_t)b * H + j] + sigm(si) * tanhf(sg);
        craw[b * 64 + col] = c1;
        cs += c1; cq += c1 * c1;
    }
    __syncthreads();            // red[] reuse safe
    red[tx] = cs; red[256 + tx] = cq;
    __syncthreads();
    if (tx < 64) {
        const int jj = blockIdx.x * 64 + tx;
        float s = red[tx] + red[64+tx] + red[128+tx] + red[192+tx];
        float q = red[256+tx] + red[256+64+tx] + red[256+128+tx] + red[256+192+tx];
        float m  = s * invB;
        float iv = rsqrtf(fmaxf(q * invB - m * m, 0.f) + EPS);
        float a  = g_c[jj] * iv;
        coef[8*64+tx] = a;
        coef[9*64+tx] = bt_c[jj] - a * m;
    }
    __syncthreads();
    const float ac = coef[8*64+col], bc = coef[9*64+col];

    // ---- phase 3: h update + length masking + outputs
    for (int bb = 0; bb < 16; ++bb) {
        const int b = b0 + bb;
        const size_t ro = (size_t)b * C4;
        const size_t ho = (size_t)b * H + j;
        float c1 = craw[b * 64 + col];
        float so = ao * (sA[ro + co] + sB[ro + co]) + bo_ + bf2f(bn_ih[ro + co]);
        float h1 = sigm(so) * tanhf(ac * c1 + bc);
        bool  m  = t < length[b];
        float hn = m ? h1 : h[ho];
        float cn = m ? c1 : c[ho];
        h[ho] = hn; c[ho] = cn; hb[ho] = f2bf_bits(hn);
        const size_t yo = (size_t)t * B * H + ho;
        if (y32) y32[yo] = hn;
        if (y16) y16[yo] = f2bf_bits(hn);
    }
}

// ---------------------------------------------------------------------------
// Host launcher
// ---------------------------------------------------------------------------
static inline void launch_gemm(const unsigned short* A, const unsigned short* Bt, void* C,
                               int M, int N, int K, int split, bool outBf16, hipStream_t s) {
    int strips = (M / 16) * (N / 64) * split;
    int blocks = (strips + 7) / 8;       // 8 waves (256 threads) per block
    if (outBf16) gemm_wmma_bf16<true ><<<blocks, 256, 0, s>>>(A, Bt, C, M, N, K, split);
    else         gemm_wmma_bf16<false><<<blocks, 256, 0, s>>>(A, Bt, C, M, N, K, split);
}

extern "C" void kernel_launch(void* const* d_in, const int* in_sizes, int n_in,
                              void* d_out, int out_size, void* d_ws, size_t ws_size,
                              hipStream_t stream) {
    const int T = 256, B = 64, D = 512, H = 1024, C4 = 4096;
    const int M = T * B;

    const float* x      = (const float*)d_in[0];
    const int*   length = (const int*)  d_in[1];
    const float* w_ih[2] = { (const float*)d_in[2],  (const float*)d_in[11] };
    const float* w_hh[2] = { (const float*)d_in[3],  (const float*)d_in[12] };
    const float* bia[2]  = { (const float*)d_in[4],  (const float*)d_in[13] };
    const float* g_ih[2] = { (const float*)d_in[5],  (const float*)d_in[14] };
    const float* bt_ih[2]= { (const float*)d_in[6],  (const float*)d_in[15] };
    const float* g_hh[2] = { (const float*)d_in[7],  (const float*)d_in[16] };
    const float* bt_hh[2]= { (const float*)d_in[8],  (const float*)d_in[17] };
    const float* g_c[2]  = { (const float*)d_in[9],  (const float*)d_in[18] };
    const float* bt_c[2] = { (const float*)d_in[10], (const float*)d_in[19] };

    char* ws = (char*)d_ws; size_t off = 0;
    auto alloc = [&](size_t bytes) -> void* {
        void* p = ws + off; off = (off + bytes + 255) & ~(size_t)255; return p;
    };
    unsigned short* S     = (unsigned short*)alloc((size_t)M * C4 * 2);  // shared by both layers
    unsigned short* xb    = (unsigned short*)alloc((size_t)M * D * 2);
    unsigned short* y0b   = (unsigned short*)alloc((size_t)M * H * 2);
    unsigned short* wihT[2] = { (unsigned short*)alloc((size_t)C4 * D * 2),
                                (unsigned short*)alloc((size_t)C4 * H * 2) };
    unsigned short* whhT[2] = { (unsigned short*)alloc((size_t)C4 * H * 2),
                                (unsigned short*)alloc((size_t)C4 * H * 2) };
    float* s_hh = (float*)alloc((size_t)2 * B * C4 * 4);   // two split-K partials
    float* h    = (float*)alloc((size_t)B * H * 4);
    float* c    = (float*)alloc((size_t)B * H * 4);
    unsigned short* hb = (unsigned short*)alloc((size_t)B * H * 2);

    float* out    = (float*)d_out;
    float* y1     = out;
    float* hn_out = out + (size_t)T * B * H;
    float* cn_out = hn_out + (size_t)2 * B * H;

    // one-time conversions / weight transposes to bf16
    {
        size_t n = (size_t)M * D;
        cvt_f32_bf16<<<(unsigned)((n + 255) / 256), 256, 0, stream>>>(x, xb, n);
        transpose_f32_bf16<<<(unsigned)(((size_t)D * C4 + 255) / 256), 256, 0, stream>>>(w_ih[0], wihT[0], D, C4);
        transpose_f32_bf16<<<(unsigned)(((size_t)H * C4 + 255) / 256), 256, 0, stream>>>(w_ih[1], wihT[1], H, C4);
        transpose_f32_bf16<<<(unsigned)(((size_t)H * C4 + 255) / 256), 256, 0, stream>>>(w_hh[0], whhT[0], H, C4);
        transpose_f32_bf16<<<(unsigned)(((size_t)H * C4 + 255) / 256), 256, 0, stream>>>(w_hh[1], whhT[1], H, C4);
    }

    for (int L = 0; L < 2; ++L) {
        // hoisted input projection for all T, then in-place BN over batch per (t, col)
        if (L == 0) launch_gemm(xb,  wihT[0], S, M, C4, D, 1, /*bf16 out*/true, stream);
        else        launch_gemm(y0b, wihT[1], S, M, C4, H, 1, true, stream);
        bn_cols_bf16<<<(unsigned)(((size_t)T * C4 + 255) / 256), 256, 0, stream>>>(
            S, g_ih[L], bt_ih[L], B, C4, (size_t)T * C4);

        zero_state<<<(B * H + 255) / 256, 256, 0, stream>>>(h, c, hb, B * H);

        for (int t = 0; t < T; ++t) {
            // recurrent GEMM: (64 x 1024) @ (1024 x 4096), bf16 WMMA,
            // split-K=2 -> two f32 partial buffers, 512 waves in flight
            launch_gemm(hb, whhT[L], s_hh, B, C4, H, 2, /*f32 out*/false, stream);
            lstm_pointwise<<<H / 64, 256, 0, stream>>>(
                s_hh, s_hh + (size_t)B * C4,
                S + (size_t)t * B * C4, bia[L], g_hh[L], bt_hh[L], g_c[L], bt_c[L],
                length, t, B, H, h, c, hb,
                (L == 1) ? y1 : nullptr, (L == 0) ? y0b : nullptr);
        }

        hipMemcpyAsync(hn_out + (size_t)L * B * H, h, (size_t)B * H * 4,
                       hipMemcpyDeviceToDevice, stream);
        hipMemcpyAsync(cn_out + (size_t)L * B * H, c, (size_t)B * H * 4,
                       hipMemcpyDeviceToDevice, stream);
    }
}
